// RMAC_33440615366843
// MI455X (gfx1250) — matrix operand order repaired and checked by reading, not compiled, not run on hardware
//
#include <hip/hip_runtime.h>

typedef __attribute__((ext_vector_type(16))) _Float16 v16h;
typedef __attribute__((ext_vector_type(8)))  _Float16 v8h;
typedef __attribute__((ext_vector_type(4)))  _Float16 v4h;
typedef __attribute__((ext_vector_type(8)))  float    v8f;

#define B_      512
#define Hh      14
#define Ww      14
#define Cc      512
#define NREG    14
#define NBOX    70   // 14 regions * (1 + 4)
#define M_TOT   (B_ * NBOX)   // 35840

// Per-region split bounds {y1, ym, y2, x1, xm, x2}, replicating the numpy
// _rmac_regions + _crop_boxes rounding (banker's rounding) exactly.
__device__ __constant__ int REG6[NREG][6] = {
    {0, 7, 14,  0, 7, 14},
    {0, 4, 9,   0, 4, 9},
    {0, 4, 9,   5, 10, 14},
    {5, 10, 14, 0, 4, 9},
    {5, 10, 14, 5, 10, 14},
    {0, 4, 7,   0, 4, 7},
    {0, 4, 7,   3, 6, 10},
    {0, 4, 7,   7, 10, 14},
    {3, 6, 10,  0, 4, 7},
    {3, 6, 10,  3, 6, 10},
    {3, 6, 10,  7, 10, 14},
    {7, 10, 14, 0, 4, 7},
    {7, 10, 14, 3, 6, 10},
    {7, 10, 14, 7, 10, 14},
};

__device__ inline float scanmax(const float* lds, int c, int ya, int yb, int xa, int xb) {
    float m = -__builtin_inff();
    for (int i = ya; i < yb; ++i) {
        const float* row = lds + (i * Ww) * 128 + c;
        for (int j = xa; j < xb; ++j)
            m = fmaxf(m, row[j * 128]);
    }
    return m;
}

// Kernel 1: per-batch ROI max pooling. LDS-stage x[b] in 128-channel chunks
// (196*128*4 = 100KB of the 320KB/WGP), scan the 56 leaf boxes once, derive
// the 14 p=1 boxes as max of their leaves. Output f16 pooled matrix (M_TOT x C).
__global__ void rmac_pool(const float* __restrict__ x, _Float16* __restrict__ A) {
    extern __shared__ float lds[];          // 196 * 128 floats
    const int b = blockIdx.x;
    const int tid = threadIdx.x;            // 128 threads
    for (int chunk = 0; chunk < 4; ++chunk) {
        __syncthreads();
        const float* xb = x + ((size_t)b * (Hh * Ww)) * Cc + chunk * 128;
        for (int idx = tid; idx < Hh * Ww * 32; idx += 128) {
            const int hw = idx >> 5, c4 = idx & 31;
            const float4 v = *(const float4*)(xb + (size_t)hw * Cc + c4 * 4);
            *(float4*)(lds + hw * 128 + c4 * 4) = v;
        }
        __syncthreads();
        const int c = tid;
        for (int r = 0; r < NREG; ++r) {
            const int y1 = REG6[r][0], ym = REG6[r][1], y2 = REG6[r][2];
            const int x1 = REG6[r][3], xm = REG6[r][4], x2 = REG6[r][5];
            // leaf order matches reference: (ix,jy) = (0,0),(0,1),(1,0),(1,1)
            const float m00 = scanmax(lds, c, y1, ym, x1, xm);
            const float m01 = scanmax(lds, c, ym, y2, x1, xm);
            const float m10 = scanmax(lds, c, y1, ym, xm, x2);
            const float m11 = scanmax(lds, c, ym, y2, xm, x2);
            const float p1 = fmaxf(fmaxf(m00, m01), fmaxf(m10, m11));
            size_t base = ((size_t)(b * NBOX + r * 5)) * Cc + chunk * 128 + c;
            A[base]            = (_Float16)p1;
            A[base + 1 * Cc]   = (_Float16)m00;
            A[base + 2 * Cc]   = (_Float16)m01;
            A[base + 3 * Cc]   = (_Float16)m10;
            A[base + 4 * Cc]   = (_Float16)m11;
        }
    }
}

__device__ inline float blockAllReduce(float v, float* sh) {
    for (int m = 16; m; m >>= 1) v += __shfl_xor(v, m, 32);
    const int w = threadIdx.x >> 5;
    const int nw = (blockDim.x + 31) >> 5;
    if ((threadIdx.x & 31) == 0) sh[w] = v;
    __syncthreads();
    float t = 0.f;
    for (int i = 0; i < nw; ++i) t += sh[i];
    __syncthreads();
    return t;
}

// Kernel 2: in-place L2 normalize each 512-element f16 row (one block per row).
__global__ void rmac_normrows(_Float16* __restrict__ A) {
    __shared__ float sh[8];
    _Float16* p = A + (size_t)blockIdx.x * Cc;
    const int tid = threadIdx.x;            // 128 threads * 4 elems
    v4h v = *(v4h*)(p + tid * 4);
    const float f0 = (float)v[0], f1 = (float)v[1], f2 = (float)v[2], f3 = (float)v[3];
    const float s = blockAllReduce(f0 * f0 + f1 * f1 + f2 * f2 + f3 * f3, sh);
    const float inv = 1.0f / fmaxf(sqrtf(s), 1e-12f);
    v[0] = (_Float16)(f0 * inv); v[1] = (_Float16)(f1 * inv);
    v[2] = (_Float16)(f2 * inv); v[3] = (_Float16)(f3 * inv);
    *(v4h*)(p + tid * 4) = v;
}

// Kernel 3: convert pca_w fp32 -> f16 once (then fully L2-resident for GEMM).
__global__ void rmac_cvtw(const float* __restrict__ w, _Float16* __restrict__ wh) {
    const int i = blockIdx.x * blockDim.x + threadIdx.x;
    wh[i] = (_Float16)w[i];
}

__device__ inline v16h frag16(v8h lo, v8h hi) {
    v16h r;
#pragma unroll
    for (int i = 0; i < 8; ++i) { r[i] = lo[i]; r[i + 8] = hi[i]; }
    return r;
}

// Kernel 4: GEMM out[m,d] = sum_c A[m,c] * pca_w[d,c] + bias[d], via
// v_wmma_f32_16x16x32_f16. Block = 8 waves (256 thr) tiling 128(M) x 64(N);
// wave tile = 32x32 (2x2 fragments). K = 512 -> 16 wmma steps per fragment.
__global__ void rmac_gemm(const _Float16* __restrict__ A, const _Float16* __restrict__ W,
                          const float* __restrict__ bias, _Float16* __restrict__ Out) {
    const int wave = threadIdx.x >> 5;
    const int lane = threadIdx.x & 31;
    const int wm = wave & 3, wn = wave >> 2;
    const int m0 = blockIdx.x * 128 + wm * 32;
    const int n0 = blockIdx.y * 64 + wn * 32;
    const int lm = lane & 15;
    const int lhalf = lane >> 4;
    const int kloA = lhalf * 8;    // A 16x32 f16: lanes 0-15 hold K 0-7 & 16-23; 16-31 hold 8-15 & 24-31
    const int kloB = lhalf * 8;    // B 32x16 f16: lanes 0-15 hold K 0-7 & 16-23 of col N; 16-31 hold 8-15 & 24-31

    v8f acc[2][2] = {};
    for (int kk = 0; kk < Cc; kk += 32) {
        v16h a[2], bf[2];
#pragma unroll
        for (int im = 0; im < 2; ++im) {
            const _Float16* pa = A + (size_t)(m0 + im * 16 + lm) * Cc + kk + kloA;
            a[im] = frag16(*(const v8h*)pa, *(const v8h*)(pa + 16));
        }
#pragma unroll
        for (int in = 0; in < 2; ++in) {
            const _Float16* pb = W + (size_t)(n0 + in * 16 + lm) * Cc + kk + kloB;
            bf[in] = frag16(*(const v8h*)pb, *(const v8h*)(pb + 16));
        }
#pragma unroll
        for (int im = 0; im < 2; ++im)
#pragma unroll
            for (int in = 0; in < 2; ++in)
                acc[im][in] = __builtin_amdgcn_wmma_f32_16x16x32_f16(
                    false, a[im], false, bf[in], (short)0, acc[im][in], false, false);
    }
    // Epilogue: + bias, store f16. C/D layout: vgpr v, lanes 0-15 -> M=v, 16-31 -> M=v+8; N=lane%16.
#pragma unroll
    for (int im = 0; im < 2; ++im)
#pragma unroll
        for (int in = 0; in < 2; ++in) {
            const int d = n0 + in * 16 + lm;
            const float bv = bias[d];
#pragma unroll
            for (int v = 0; v < 8; ++v) {
                const int m = m0 + im * 16 + v + lhalf * 8;
                Out[(size_t)m * Cc + d] = (_Float16)(acc[im][in][v] + bv);
            }
        }
}

// Kernel 5: per batch: normalize each of the 70 GEMM rows, sum them, final L2
// normalize, write fp32 output. One block (128 thr, 4 cols each) per batch.
__global__ void rmac_finalize(const _Float16* __restrict__ G, float* __restrict__ out) {
    __shared__ float sh[8];
    const int b = blockIdx.x, tid = threadIdx.x;
    float s0 = 0.f, s1 = 0.f, s2 = 0.f, s3 = 0.f;
    for (int r = 0; r < NBOX; ++r) {
        const _Float16* p = G + ((size_t)(b * NBOX + r)) * Cc + tid * 4;
        const v4h v = *(const v4h*)p;
        const float f0 = (float)v[0], f1 = (float)v[1], f2 = (float)v[2], f3 = (float)v[3];
        const float ss = blockAllReduce(f0 * f0 + f1 * f1 + f2 * f2 + f3 * f3, sh);
        const float inv = 1.0f / fmaxf(sqrtf(ss), 1e-12f);
        s0 += f0 * inv; s1 += f1 * inv; s2 += f2 * inv; s3 += f3 * inv;
    }
    const float ss = blockAllReduce(s0 * s0 + s1 * s1 + s2 * s2 + s3 * s3, sh);
    const float inv = 1.0f / fmaxf(sqrtf(ss), 1e-12f);
    float* o = out + (size_t)b * Cc + tid * 4;
    o[0] = s0 * inv; o[1] = s1 * inv; o[2] = s2 * inv; o[3] = s3 * inv;
}

extern "C" void kernel_launch(void* const* d_in, const int* in_sizes, int n_in,
                              void* d_out, int out_size, void* d_ws, size_t ws_size,
                              hipStream_t stream) {
    (void)in_sizes; (void)n_in; (void)out_size; (void)ws_size;
    const float* x     = (const float*)d_in[0];   // (512,14,14,512) f32
    const float* pca_w = (const float*)d_in[1];   // (512,512) f32
    const float* pca_b = (const float*)d_in[2];   // (512,) f32
    float* out = (float*)d_out;                   // (512,512) f32

    char* ws = (char*)d_ws;
    const size_t szA = (size_t)M_TOT * Cc * sizeof(_Float16);   // 36,700,160 B
    const size_t szW = (size_t)Cc * Cc * sizeof(_Float16);      //    524,288 B
    _Float16* Ah = (_Float16*)ws;
    _Float16* Wh = (_Float16*)(ws + szA);
    _Float16* Gh = (_Float16*)(ws + szA + szW);

    rmac_cvtw<<<1024, 256, 0, stream>>>(pca_w, Wh);
    rmac_pool<<<B_, 128, Hh * Ww * 128 * sizeof(float), stream>>>(x, Ah);
    rmac_normrows<<<M_TOT, 128, 0, stream>>>(Ah);
    rmac_gemm<<<dim3(M_TOT / 128, Cc / 64), 256, 0, stream>>>(Ah, Wh, pca_b, Gh);
    rmac_finalize<<<B_, 128, 0, stream>>>(Gh, out);
}
